// GraphAttentionLayer_43215960933245
// MI455X (gfx1250) — matrix operand order repaired and checked by reading, not compiled, not run on hardware
//
#include <hip/hip_runtime.h>

#define GAT_N 8192
#define GAT_D 64
#define GAT_LOG2E 1.4426950408889634f

typedef __attribute__((ext_vector_type(16))) _Float16 v16h;
typedef __attribute__((ext_vector_type(8)))  float    v8f;

// ---------------------------------------------------------------------------
// Prep: src2 = (h@a1)*log2e, dst2 = (h@a2)*log2e, hT = f16 transpose of h
// (column-major so WMMA B tiles are contiguous-in-K per lane).
// ---------------------------------------------------------------------------
__global__ __launch_bounds__(256) void gat_prep(const float* __restrict__ h,
                                                const float* __restrict__ a,
                                                float* __restrict__ src2,
                                                float* __restrict__ dst2,
                                                _Float16* __restrict__ hT) {
  int i = blockIdx.x * blockDim.x + threadIdx.x;
  if (i >= GAT_N) return;
  float s = 0.f, d = 0.f;
  #pragma unroll
  for (int c = 0; c < GAT_D; ++c) {
    float v = h[(size_t)i * GAT_D + c];
    s += v * a[c];
    d += v * a[GAT_D + c];
    hT[(size_t)c * GAT_N + i] = (_Float16)v;   // coalesced across threads per c
  }
  src2[i] = s * GAT_LOG2E;
  dst2[i] = d * GAT_LOG2E;
}

// ---------------------------------------------------------------------------
// Global max of dst2 (single workgroup).
// ---------------------------------------------------------------------------
__global__ __launch_bounds__(256) void gat_dstmax(const float* __restrict__ dst2,
                                                  float* __restrict__ dmax) {
  __shared__ float red[8];
  float m = -3.0e38f;
  for (int i = threadIdx.x; i < GAT_N; i += 256) m = fmaxf(m, dst2[i]);
  #pragma unroll
  for (int off = 16; off > 0; off >>= 1) m = fmaxf(m, __shfl_xor(m, off, 32));
  if ((threadIdx.x & 31) == 0) red[threadIdx.x >> 5] = m;
  __syncthreads();
  if (threadIdx.x == 0) {
    float v = red[0];
    #pragma unroll
    for (int w = 1; w < 8; ++w) v = fmaxf(v, red[w]);
    dmax[0] = v;
  }
}

// ---------------------------------------------------------------------------
// Per-K-block register buffer (adj in A-layout, pre-scaled dst, 4 B tiles).
// Kept as vector components only, so SROA keeps it entirely in VGPRs.
// ---------------------------------------------------------------------------
struct KBuf {
  int4   aj[4];
  float4 dv[4];
  v16h   B[4];
};

__device__ __forceinline__ void kload(KBuf& b, const int* __restrict__ arowp,
                                      const float* __restrict__ dst2,
                                      const _Float16* __restrict__ bbase,
                                      int kb, int k0, int k1) {
  b.aj[0] = *(const int4*)(arowp + kb + k0);
  b.aj[1] = *(const int4*)(arowp + kb + k0 + 4);
  b.aj[2] = *(const int4*)(arowp + kb + k1);
  b.aj[3] = *(const int4*)(arowp + kb + k1 + 4);
  b.dv[0] = *(const float4*)(dst2 + kb + k0);
  b.dv[1] = *(const float4*)(dst2 + kb + k0 + 4);
  b.dv[2] = *(const float4*)(dst2 + kb + k1);
  b.dv[3] = *(const float4*)(dst2 + kb + k1 + 4);
  const _Float16* bb = bbase + kb;
  b.B[0] = *(const v16h*)(bb);
  b.B[1] = *(const v16h*)(bb + 16 * (size_t)GAT_N);
  b.B[2] = *(const v16h*)(bb + 32 * (size_t)GAT_N);
  b.B[3] = *(const v16h*)(bb + 48 * (size_t)GAT_N);
  // pull the adj stream further ahead into L2 (gfx1250 global_prefetch_b8)
  __builtin_prefetch(arowp + kb + 64 + k0, 0, 1);
}

__device__ __forceinline__ void kcompute(const KBuf& b, float sM, float Cc,
                                         const v16h& onesB,
                                         v8f& acc0, v8f& acc1, v8f& acc2,
                                         v8f& acc3, v8f& accL) {
  v16h Am;
  #pragma unroll
  for (int q = 0; q < 4; ++q) {
    const int4   a4 = b.aj[q];
    const float4 d4 = b.dv[q];
    float u, w, arg;
    u = sM + d4.x; w = fmaf(0.01f, u, Cc); arg = fmaxf(u, w);
    arg = a4.x > 0 ? arg : -1.0e30f;
    Am[4 * q + 0] = (_Float16)__builtin_amdgcn_exp2f(arg);
    u = sM + d4.y; w = fmaf(0.01f, u, Cc); arg = fmaxf(u, w);
    arg = a4.y > 0 ? arg : -1.0e30f;
    Am[4 * q + 1] = (_Float16)__builtin_amdgcn_exp2f(arg);
    u = sM + d4.z; w = fmaf(0.01f, u, Cc); arg = fmaxf(u, w);
    arg = a4.z > 0 ? arg : -1.0e30f;
    Am[4 * q + 2] = (_Float16)__builtin_amdgcn_exp2f(arg);
    u = sM + d4.w; w = fmaf(0.01f, u, Cc); arg = fmaxf(u, w);
    arg = a4.w > 0 ? arg : -1.0e30f;
    Am[4 * q + 3] = (_Float16)__builtin_amdgcn_exp2f(arg);
  }
  acc0 = __builtin_amdgcn_wmma_f32_16x16x32_f16(false, Am, false, b.B[0], (short)0, acc0, false, false);
  acc1 = __builtin_amdgcn_wmma_f32_16x16x32_f16(false, Am, false, b.B[1], (short)0, acc1, false, false);
  acc2 = __builtin_amdgcn_wmma_f32_16x16x32_f16(false, Am, false, b.B[2], (short)0, acc2, false, false);
  acc3 = __builtin_amdgcn_wmma_f32_16x16x32_f16(false, Am, false, b.B[3], (short)0, acc3, false, false);
  accL = __builtin_amdgcn_wmma_f32_16x16x32_f16(false, Am, false, onesB,  (short)0, accL, false, false);
}

// ---------------------------------------------------------------------------
// Main: one wave per 16-row block, single pass over adj (256 MB read once).
// Static per-row softmax bound M_i = leaky(src_i + max(dst)) -> no online
// rescaling. Denominator from a 5th WMMA against an all-ones B. Two-stage
// software pipeline (ping-pong KBufs) keeps loads a full block ahead.
// ---------------------------------------------------------------------------
__global__ __launch_bounds__(256) void gat_main(const int* __restrict__ adj,
                                                const float* __restrict__ src2,
                                                const float* __restrict__ dst2,
                                                const float* __restrict__ dmax,
                                                const _Float16* __restrict__ hT,
                                                float* __restrict__ out) {
  const int lane    = threadIdx.x & 31;
  const int wid     = blockIdx.x * (blockDim.x >> 5) + (threadIdx.x >> 5);
  const int rowbase = wid * 16;
  const int ra = lane & 15;   // A-matrix row (lanes L and L+16 share a row)
  const int hi = lane >> 4;   // lane half selects K subset (16-bit A layout)
  const int row = rowbase + ra;
  const int k0 = hi * 16;     // K runs {0..15}/{16..31} (contiguous per lane half)
  const int k1 = k0 + 8;

  const float srow2 = src2[row];
  const float dm    = dmax[0];
  const float t0    = srow2 + dm;
  const float M2    = fmaxf(t0, 0.01f * t0);    // log2-domain row upper bound
  const float sM    = srow2 - M2;
  const float Cc    = -0.99f * M2;

  v8f acc0 = {}, acc1 = {}, acc2 = {}, acc3 = {};  // D=64 -> 4 N-tiles of 16
  v8f accL = {};                                   // softmax denominator tile

  v16h onesB;
  #pragma unroll
  for (int n = 0; n < 16; ++n) onesB[n] = (_Float16)1.0f;

  const int*      arowp = adj + (size_t)row * GAT_N;
  const _Float16* bbase = hT + (size_t)ra * GAT_N + hi * 16;

  KBuf b0, b1;
  kload(b0, arowp, dst2, bbase, 0, k0, k1);
  int kb = 0;
  for (; kb < GAT_N - 64; kb += 64) {
    kload(b1, arowp, dst2, bbase, kb + 32, k0, k1);
    kcompute(b0, sM, Cc, onesB, acc0, acc1, acc2, acc3, accL);
    kload(b0, arowp, dst2, bbase, kb + 64, k0, k1);
    kcompute(b1, sM, Cc, onesB, acc0, acc1, acc2, acc3, accL);
  }
  kload(b1, arowp, dst2, bbase, kb + 32, k0, k1);
  kcompute(b0, sM, Cc, onesB, acc0, acc1, acc2, acc3, accL);
  kcompute(b1, sM, Cc, onesB, acc0, acc1, acc2, acc3, accL);

  // accL columns are all identical = row denominator, already in C layout.
  #pragma unroll
  for (int r = 0; r < 8; ++r) {
    float linv = __builtin_amdgcn_rcpf(accL[r]);
    int orow = rowbase + (hi << 3) + r;
    float* op = out + (size_t)orow * GAT_D + ra;
    op[0]  = acc0[r] * linv;
    op[16] = acc1[r] * linv;
    op[32] = acc2[r] * linv;
    op[48] = acc3[r] * linv;
  }
}

// ---------------------------------------------------------------------------
extern "C" void kernel_launch(void* const* d_in, const int* in_sizes, int n_in,
                              void* d_out, int out_size, void* d_ws, size_t ws_size,
                              hipStream_t stream) {
  const float* h   = (const float*)d_in[0];   // (N, D) f32
  const int*   adj = (const int*)d_in[1];     // (N, N) i32
  const float* a   = (const float*)d_in[2];   // (2D, 1) f32
  float* out = (float*)d_out;                 // (N, D) f32

  char* ws = (char*)d_ws;
  float*    src2 = (float*)ws;                       // 32 KB
  float*    dst2 = (float*)(ws + 32 * 1024);         // 32 KB
  _Float16* hT   = (_Float16*)(ws + 64 * 1024);      // 1 MB (64 x 8192 f16)
  float*    dmax = (float*)(ws + 64 * 1024 + (size_t)GAT_D * GAT_N * 2);

  gat_prep<<<GAT_N / 256, 256, 0, stream>>>(h, a, src2, dst2, hT);
  gat_dstmax<<<1, 256, 0, stream>>>(dst2, dmax);
  // 512 row-blocks, 8 waves (one row-block each) per 256-thread workgroup
  gat_main<<<(GAT_N / 16) / 8, 256, 0, stream>>>(adj, src2, dst2, dmax, hT, out);
}
